// DEC_HOGNN_11596411699257
// MI455X (gfx1250) — compile-verified, matrix operand order checked by reading, not run on hardware
//
#include <hip/hip_runtime.h>
#include <hip/hip_bf16.h>

// ---------------------------------------------------------------------------
// DEC_HOGNN on gfx1250: all dense GEMMs via v_wmma_f32_16x16x32_f16 with
// 4-way N register blocking; attention softmax/scatter via f32 VALU +
// hardware global atomics (atomic umax segment-max, atomic f32 scatter-add).
// ---------------------------------------------------------------------------

typedef __attribute__((ext_vector_type(16))) _Float16 v16h;
typedef __attribute__((ext_vector_type(8)))  _Float16 v8h;
typedef __attribute__((ext_vector_type(8)))  float    v8f;

union V16H { v16h v; v8h h[2]; };

#define NHEADS 3
#define DMODEL 128
#define HC     384   // H*C
#define KEE    277
#define KEEP   288   // padded to multiple of 32
#define GEO    20

// ---------------------------------------------------------------------------
// WMMA GEMM: C[M,Nout] = A[M,K](f16) @ B[K,Nout] + bias
// B pre-transposed as Bt[Nout][Kpad] (f16) so all fragments are contiguous
// 16B loads.  One wave computes a 16x64 output slab (4 accumulators): one
// A fragment is reused by 4 WMMAs per k-step -> 2.5 b128 loads per WMMA.
// Requires M%16==0, K%32==0, Nout%64==0 (true for all shapes here).
// ---------------------------------------------------------------------------
template <bool OUT_HALF>
__global__ void wmma_gemm(const _Float16* __restrict__ A, int lda,
                          const _Float16* __restrict__ Bt, int ldb,
                          const float* __restrict__ bias,
                          void* __restrict__ Cout, int ldc,
                          int M, int K)
{
    const int lane = threadIdx.x;          // 0..31
    const int r    = lane & 15;
    const int hi   = lane >> 4;
    const int m0   = blockIdx.x * 16;
    const int n0   = blockIdx.y * 64;
    if (m0 >= M) return;                   // uniform per wave

    const _Float16* arow = A  + (size_t)(m0 + r) * lda;
    const _Float16* brow = Bt + (size_t)(n0 + r) * ldb;
    const size_t bstep = (size_t)16 * ldb;

    v8f acc0 = {}, acc1 = {}, acc2 = {}, acc3 = {};
    for (int k0 = 0; k0 < K; k0 += 32) {
        if (k0 + 32 < K) {
            __builtin_prefetch(arow + k0 + 32, 0, 3);
            __builtin_prefetch(brow + k0 + 32, 0, 3);
            __builtin_prefetch(brow + 2 * bstep + k0 + 32, 0, 3);
        }
        // documented 16-bit A 16x32 striping: lanes0-15 hold K {0-7,16-23},
        // lanes16-31 hold K {8-15,24-31}; B mirrored with N = lane&15.
        V16H a, b0, b1, b2, b3;
        const _Float16* ap = arow + k0 + 8 * hi;
        const _Float16* bp = brow + k0 + 8 * hi;
        a.h[0]  = *(const v8h*)(ap);
        a.h[1]  = *(const v8h*)(ap + 16);
        b0.h[0] = *(const v8h*)(bp);
        b0.h[1] = *(const v8h*)(bp + 16);
        b1.h[0] = *(const v8h*)(bp + bstep);
        b1.h[1] = *(const v8h*)(bp + bstep + 16);
        b2.h[0] = *(const v8h*)(bp + 2 * bstep);
        b2.h[1] = *(const v8h*)(bp + 2 * bstep + 16);
        b3.h[0] = *(const v8h*)(bp + 3 * bstep);
        b3.h[1] = *(const v8h*)(bp + 3 * bstep + 16);
        acc0 = __builtin_amdgcn_wmma_f32_16x16x32_f16(
            false, a.v, false, b0.v, (short)0, acc0, false, false);
        acc1 = __builtin_amdgcn_wmma_f32_16x16x32_f16(
            false, a.v, false, b1.v, (short)0, acc1, false, false);
        acc2 = __builtin_amdgcn_wmma_f32_16x16x32_f16(
            false, a.v, false, b2.v, (short)0, acc2, false, false);
        acc3 = __builtin_amdgcn_wmma_f32_16x16x32_f16(
            false, a.v, false, b3.v, (short)0, acc3, false, false);
    }

    // C/D layout: VGPR i -> M = i + 8*hi, N = lane&15 (per 16x16 tile)
    const v8f accs[4] = {acc0, acc1, acc2, acc3};
    #pragma unroll
    for (int t = 0; t < 4; ++t) {
        const int nc = n0 + 16 * t + r;
        const float bn = bias ? bias[nc] : 0.0f;
        if (OUT_HALF) {
            _Float16* C = (_Float16*)Cout;
            #pragma unroll
            for (int i = 0; i < 8; ++i)
                C[(size_t)(m0 + i + 8 * hi) * ldc + nc] = (_Float16)(accs[t][i] + bn);
        } else {
            float* C = (float*)Cout;
            #pragma unroll
            for (int i = 0; i < 8; ++i)
                C[(size_t)(m0 + i + 8 * hi) * ldc + nc] = accs[t][i] + bn;
        }
    }
}

// ---------------------------------------------------------------------------
// Weight transpose + f32->f16 cast with K padding: Wt[n][k] = W[k][n]
// ---------------------------------------------------------------------------
__global__ void transpose_cast(const float* __restrict__ W, _Float16* __restrict__ Wt,
                               int K, int Nout, int Kpad)
{
    int idx = blockIdx.x * blockDim.x + threadIdx.x;
    if (idx >= Nout * Kpad) return;
    int n = idx / Kpad, k = idx - n * Kpad;
    Wt[idx] = (k < K) ? (_Float16)W[(size_t)k * Nout + n] : (_Float16)0.0f;
}

__global__ void cast_f16(const float* __restrict__ s, _Float16* __restrict__ d, int n)
{
    int i = blockIdx.x * blockDim.x + threadIdx.x;
    if (i < n) d[i] = (_Float16)s[i];
}

// write 128-wide f32 rows into the 256-wide f16 concat matrix
__global__ void cast_to_cat(const float* __restrict__ s, _Float16* __restrict__ cat,
                            int N, int off)
{
    int i = blockIdx.x * blockDim.x + threadIdx.x;
    if (i >= N * DMODEL) return;
    int n = i >> 7, c = i & 127;
    cat[(size_t)n * 256 + off + c] = (_Float16)s[i];
}

// ---------------------------------------------------------------------------
// Edge-encoder feature gather: F[e] = [h[dst] | h[src]*dir | dir | geo | 0pad]
// as f16, padded to 288 columns.  One block (128 threads) per edge.
// ---------------------------------------------------------------------------
__global__ void build_edge_feat(const float* __restrict__ h,
                                const float* __restrict__ geo,
                                const float* __restrict__ dirw,
                                const int* __restrict__ src,
                                const int* __restrict__ dst,
                                _Float16* __restrict__ F, int E)
{
    int e = blockIdx.x;
    if (e >= E) return;
    int c = threadIdx.x;                     // 0..127
    float d = dirw[e];
    int s = src[e], t = dst[e];
    _Float16* row = F + (size_t)e * KEEP;
    row[c]          = (_Float16)h[(size_t)t * DMODEL + c];
    row[DMODEL + c] = (_Float16)(h[(size_t)s * DMODEL + c] * d);
    if (c == 0)            row[256] = (_Float16)d;
    if (c >= 1 && c <= GEO) row[256 + c] = (_Float16)geo[(size_t)e * GEO + (c - 1)];
    if (c > GEO && c < 32)  row[256 + c] = (_Float16)0.0f;   // pad 277..287
}

// ---------------------------------------------------------------------------
// Attention: monotone float<->uint key for hardware atomic segment-max
// ---------------------------------------------------------------------------
__device__ __forceinline__ unsigned fkey(float f) {
    unsigned u = __float_as_uint(f);
    return (u & 0x80000000u) ? ~u : (u | 0x80000000u);
}
__device__ __forceinline__ float funkey(unsigned k) {
    return __uint_as_float((k & 0x80000000u) ? (k ^ 0x80000000u) : ~k);
}
#define NEG_INF_KEY 0x007FFFFFu   // fkey(-inf)

__global__ void init_attn(unsigned* __restrict__ amaxkey, float* __restrict__ den,
                          float* __restrict__ accum, int N)
{
    int i = blockIdx.x * blockDim.x + threadIdx.x;
    if (i < N * NHEADS) { amaxkey[i] = NEG_INF_KEY; den[i] = 0.0f; }
    if (i < N * HC)     accum[i] = 0.0f;
}

// alpha[e][h] = dot(q[dst], k[src]+e) / sqrt(C); atomic segment max.
// One wave per edge; 8 edges per block.
__global__ void alpha_kernel(const float* __restrict__ q, const float* __restrict__ k,
                             const float* __restrict__ ebuf,
                             const int* __restrict__ src, const int* __restrict__ dst,
                             float* __restrict__ alphab, unsigned* __restrict__ amaxkey,
                             int E)
{
    int e = blockIdx.x * blockDim.y + threadIdx.y;
    if (e >= E) return;
    int lane = threadIdx.x;
    int s = src[e], t = dst[e];
    const float* qr = q    + (size_t)t * HC;
    const float* kr = k    + (size_t)s * HC;
    const float* er = ebuf + (size_t)e * HC;
    #pragma unroll
    for (int hh = 0; hh < NHEADS; ++hh) {
        float p = 0.0f;
        #pragma unroll
        for (int tt = 0; tt < 4; ++tt) {
            int idx = hh * DMODEL + lane + 32 * tt;
            p += qr[idx] * (kr[idx] + er[idx]);
        }
        for (int off = 16; off; off >>= 1) p += __shfl_xor(p, off, 32);
        if (lane == 0) {
            float a = p * 0.08838834764831843f;   // 1/sqrt(128)
            alphab[(size_t)e * NHEADS + hh] = a;
            atomicMax(&amaxkey[(size_t)t * NHEADS + hh], fkey(a));
        }
    }
}

// ex = exp(alpha - amax[dst]); den[dst] += ex   (alpha buffer overwritten)
__global__ void softmax_ex(const int* __restrict__ dst, float* __restrict__ alphab,
                           const unsigned* __restrict__ amaxkey, float* __restrict__ den,
                           int E)
{
    int i = blockIdx.x * blockDim.x + threadIdx.x;
    if (i >= E * NHEADS) return;
    int e = i / NHEADS, hh = i - e * NHEADS;
    int t = dst[e];
    unsigned kk = amaxkey[(size_t)t * NHEADS + hh];
    float m = (kk == NEG_INF_KEY) ? 0.0f : funkey(kk);
    float ex = expf(alphab[i] - m);
    alphab[i] = ex;
    atomicAdd(&den[(size_t)t * NHEADS + hh], ex);
}

// accum[dst] += (v[src] + e) * ex/(den[dst]+1e-16).  One block per edge.
__global__ void scatter_msg(const float* __restrict__ v, const float* __restrict__ ebuf,
                            const float* __restrict__ alphab, const float* __restrict__ den,
                            const int* __restrict__ src, const int* __restrict__ dst,
                            float* __restrict__ accum, int E)
{
    int e = blockIdx.x;
    if (e >= E) return;
    int c = threadIdx.x;                 // 0..127
    int s = src[e], t = dst[e];
    #pragma unroll
    for (int hh = 0; hh < NHEADS; ++hh) {
        float w = alphab[(size_t)e * NHEADS + hh] /
                  (den[(size_t)t * NHEADS + hh] + 1e-16f);
        int idx = hh * DMODEL + c;
        float val = (v[(size_t)s * HC + idx] + ebuf[(size_t)e * HC + idx]) * w;
        atomicAdd(&accum[(size_t)t * HC + idx], val);
    }
}

// h_new = relu(head_mean(accum) + skip + residual); also recast to f16
__global__ void finalize_layer(const float* __restrict__ accum,
                               const float* __restrict__ skip,
                               const float* __restrict__ residual,
                               float* __restrict__ hbuf, _Float16* __restrict__ h16,
                               int N)
{
    int i = blockIdx.x * blockDim.x + threadIdx.x;
    if (i >= N * DMODEL) return;
    int n = i >> 7, c = i & 127;
    const float* ar = accum + (size_t)n * HC;
    float m = (ar[c] + ar[DMODEL + c] + ar[2 * DMODEL + c]) * (1.0f / 3.0f);
    float o = m + skip[i] + residual[i];
    o = o > 0.0f ? o : 0.0f;
    hbuf[i] = o;
    h16[i]  = (_Float16)o;
}

// ---------------------------------------------------------------------------

extern "C" void kernel_launch(void* const* d_in, const int* in_sizes, int n_in,
                              void* d_out, int out_size, void* d_ws, size_t ws_size,
                              hipStream_t stream)
{
    const float* h     = (const float*)d_in[0];
    const float* geo   = (const float*)d_in[1];
    const float* dirw  = (const float*)d_in[2];
    const int*   src   = (const int*)d_in[3];
    const int*   dst   = (const int*)d_in[4];
    const float* W_ee  = (const float*)d_in[5];
    const float* b_ee  = (const float*)d_in[6];
    const float* W_res = (const float*)d_in[7];
    const float* b_res = (const float*)d_in[8];
    const float* Wq    = (const float*)d_in[9];
    const float* bq    = (const float*)d_in[10];
    const float* Wk    = (const float*)d_in[11];
    const float* bk    = (const float*)d_in[12];
    const float* Wv    = (const float*)d_in[13];
    const float* bv    = (const float*)d_in[14];
    const float* We    = (const float*)d_in[15];
    const float* be    = (const float*)d_in[16];
    const float* Ws    = (const float*)d_in[17];
    const float* bs    = (const float*)d_in[18];
    const float* Wc    = (const float*)d_in[19];
    const float* bc    = (const float*)d_in[20];

    const int N = in_sizes[0] / DMODEL;   // 50000
    const int E = in_sizes[3];            // 400000

    // ---- bump-allocate workspace (256B aligned) ----
    char* p = (char*)d_ws;
    auto alloc = [&](size_t bytes) -> void* {
        void* r = (void*)p;
        p += (bytes + 255) & ~(size_t)255;
        return r;
    };
    _Float16* h16     = (_Float16*)alloc((size_t)N * DMODEL * 2);
    _Float16* cat16   = (_Float16*)alloc((size_t)N * 256 * 2);
    _Float16* F16     = (_Float16*)alloc((size_t)E * KEEP * 2);
    _Float16* eattr16 = (_Float16*)alloc((size_t)E * DMODEL * 2);
    float*    qbuf    = (float*)alloc((size_t)N * HC * 4);
    float*    kbuf    = (float*)alloc((size_t)N * HC * 4);
    float*    vbuf    = (float*)alloc((size_t)N * HC * 4);
    float*    ebuf    = (float*)alloc((size_t)E * HC * 4);
    float*    skipb   = (float*)alloc((size_t)N * DMODEL * 4);
    float*    resb    = (float*)alloc((size_t)N * DMODEL * 4);
    float*    hbuf    = (float*)alloc((size_t)N * DMODEL * 4);
    float*    alphab  = (float*)alloc((size_t)E * NHEADS * 4);
    unsigned* amaxkey = (unsigned*)alloc((size_t)N * NHEADS * 4);
    float*    den     = (float*)alloc((size_t)N * NHEADS * 4);
    float*    accum   = (float*)alloc((size_t)N * HC * 4);
    _Float16* Wt_ee   = (_Float16*)alloc((size_t)DMODEL * KEEP * 2);
    _Float16* Wt_res  = (_Float16*)alloc((size_t)DMODEL * DMODEL * 2);
    _Float16* Wt_q[3], *Wt_k[3], *Wt_v[3], *Wt_e[3], *Wt_s[3];
    for (int i = 0; i < 3; ++i) {
        Wt_q[i] = (_Float16*)alloc((size_t)HC * DMODEL * 2);
        Wt_k[i] = (_Float16*)alloc((size_t)HC * DMODEL * 2);
        Wt_v[i] = (_Float16*)alloc((size_t)HC * DMODEL * 2);
        Wt_e[i] = (_Float16*)alloc((size_t)HC * DMODEL * 2);
        Wt_s[i] = (_Float16*)alloc((size_t)DMODEL * DMODEL * 2);
    }
    _Float16* Wt_c = (_Float16*)alloc((size_t)DMODEL * 256 * 2);

    // ---- helpers ----
    auto tc = [&](const float* W, _Float16* Wt, int K, int Nout, int Kpad) {
        int tot = Nout * Kpad;
        transpose_cast<<<(tot + 255) / 256, 256, 0, stream>>>(W, Wt, K, Nout, Kpad);
    };
    auto gemm = [&](const _Float16* A, int lda, const _Float16* Bt, int ldb,
                    const float* bias, void* C, int ldc, int M, int K, int Nout,
                    bool half_out) {
        dim3 grid((unsigned)(M / 16), (unsigned)(Nout / 64));
        if (half_out)
            wmma_gemm<true ><<<grid, 32, 0, stream>>>(A, lda, Bt, ldb, bias, C, ldc, M, K);
        else
            wmma_gemm<false><<<grid, 32, 0, stream>>>(A, lda, Bt, ldb, bias, C, ldc, M, K);
    };

    // ---- preprocessing: casts, weight transposes, edge features ----
    cast_f16<<<(N * DMODEL + 255) / 256, 256, 0, stream>>>(h, h16, N * DMODEL);
    cast_to_cat<<<(N * DMODEL + 255) / 256, 256, 0, stream>>>(h, cat16, N, 0);
    build_edge_feat<<<E, 128, 0, stream>>>(h, geo, dirw, src, dst, F16, E);

    tc(W_ee, Wt_ee, KEE, DMODEL, KEEP);
    tc(W_res, Wt_res, DMODEL, DMODEL, DMODEL);
    for (int i = 0; i < 3; ++i) {
        tc(Wq + (size_t)i * DMODEL * HC, Wt_q[i], DMODEL, HC, DMODEL);
        tc(Wk + (size_t)i * DMODEL * HC, Wt_k[i], DMODEL, HC, DMODEL);
        tc(Wv + (size_t)i * DMODEL * HC, Wt_v[i], DMODEL, HC, DMODEL);
        tc(We + (size_t)i * DMODEL * HC, Wt_e[i], DMODEL, HC, DMODEL);
        tc(Ws + (size_t)i * DMODEL * DMODEL, Wt_s[i], DMODEL, DMODEL, DMODEL);
    }
    tc(Wc, Wt_c, 256, DMODEL, 256);

    // edge encoder GEMM -> edge_attr (f16, reused by all 3 layers)
    gemm(F16, KEEP, Wt_ee, KEEP, b_ee, eattr16, DMODEL, E, KEEP, DMODEL, true);

    // ---- 3 TransformerConv layers ----
    for (int i = 0; i < 3; ++i) {
        gemm(h16, DMODEL, Wt_q[i], DMODEL, bq + (size_t)i * HC, qbuf, HC, N, DMODEL, HC, false);
        gemm(h16, DMODEL, Wt_k[i], DMODEL, bk + (size_t)i * HC, kbuf, HC, N, DMODEL, HC, false);
        gemm(h16, DMODEL, Wt_v[i], DMODEL, bv + (size_t)i * HC, vbuf, HC, N, DMODEL, HC, false);
        gemm(eattr16, DMODEL, Wt_e[i], DMODEL, be + (size_t)i * HC, ebuf, HC, E, DMODEL, HC, false);
        gemm(h16, DMODEL, Wt_s[i], DMODEL, bs + (size_t)i * DMODEL, skipb, DMODEL, N, DMODEL, DMODEL, false);
        if (i == 0)
            gemm(h16, DMODEL, Wt_res, DMODEL, b_res, resb, DMODEL, N, DMODEL, DMODEL, false);

        init_attn<<<(N * HC + 255) / 256, 256, 0, stream>>>(amaxkey, den, accum, N);
        alpha_kernel<<<(E + 7) / 8, dim3(32, 8), 0, stream>>>(qbuf, kbuf, ebuf, src, dst,
                                                              alphab, amaxkey, E);
        softmax_ex<<<(E * NHEADS + 255) / 256, 256, 0, stream>>>(dst, alphab, amaxkey, den, E);
        scatter_msg<<<E, 128, 0, stream>>>(vbuf, ebuf, alphab, den, src, dst, accum, E);
        finalize_layer<<<(N * DMODEL + 255) / 256, 256, 0, stream>>>(
            accum, skipb, (i == 0) ? resb : hbuf, hbuf, h16, N);
    }

    // ---- output: concat([h, h_cur]) @ W_ctr + b_ctr ----
    cast_to_cat<<<(N * DMODEL + 255) / 256, 256, 0, stream>>>(hbuf, cat16, N, DMODEL);
    gemm(cat16, 256, Wt_c, 256, bc, (float*)d_out, DMODEL, N, 256, DMODEL, false);
}
